// Transformer_52527450030086
// MI455X (gfx1250) — matrix operand order repaired and checked by reading, not compiled, not run on hardware
//
#include <hip/hip_runtime.h>
#include <cstdint>
#include <cstddef>

#define USE_ASYNC 1

// ---------------- types ----------------
typedef __attribute__((ext_vector_type(16))) __bf16 v16bf;
typedef __attribute__((ext_vector_type(8)))  float  v8f;

union FragBF { unsigned int u[8]; v16bf v; };

__device__ __forceinline__ unsigned short f2bf(float f) {
    unsigned int u = __float_as_uint(f);
    u = (u + 0x7FFFu + ((u >> 16) & 1u)) >> 16;     // round-to-nearest-even
    return (unsigned short)u;
}

// 16-byte global -> LDS copy; async (ASYNCcnt-tracked) on the CDNA5 path.
__device__ __forceinline__ void copy16_g2l(const void* g, void* l) {
#if USE_ASYNC
    asm volatile("global_load_async_to_lds_b128 %0, %1, off"
                 :: "v"((unsigned int)(unsigned long long)l), "v"(g)
                 : "memory");
#else
    *(uint4*)l = *(const uint4*)g;
#endif
}
__device__ __forceinline__ void wait_async_all() {
#if USE_ASYNC
    asm volatile("s_wait_asynccnt 0x0" ::: "memory");
#endif
}

// ---------------- f32 -> bf16 convert ----------------
__global__ __launch_bounds__(256) void conv_f32_bf16(const float* __restrict__ in,
                                                     unsigned short* __restrict__ out, int n) {
    int i = blockIdx.x * 256 + threadIdx.x;
    if (i < n) out[i] = f2bf(in[i]);
}

// ---------------- embedding gather ----------------
__global__ __launch_bounds__(256) void embed_kernel(const int* __restrict__ tokens,
                                                    const int* __restrict__ positions,
                                                    const float* __restrict__ emb,
                                                    const float* __restrict__ pos_table,
                                                    float* __restrict__ x) {
    int i = blockIdx.x * 256 + threadIdx.x;       // 1024*512 total
    int node = i >> 9, d = i & 511;
    x[i] = emb[(size_t)tokens[node] * 512 + d] * 22.62741699796952f
         + pos_table[(size_t)positions[node] * 512 + d];
}

// ---------------- layernorm (f32 in -> bf16 out) ----------------
__global__ __launch_bounds__(256) void layernorm_bf16(const float* __restrict__ x,
                                                      const float* __restrict__ gamma,
                                                      const float* __restrict__ beta,
                                                      unsigned short* __restrict__ out) {
    __shared__ float sh[8];
    int row = blockIdx.x, tid = threadIdx.x;
    const float* p = x + (size_t)row * 512;
    float v0 = p[tid], v1 = p[tid + 256];
    float s = v0 + v1;
    #pragma unroll
    for (int o = 16; o; o >>= 1) s += __shfl_xor(s, o, 32);
    if ((tid & 31) == 0) sh[tid >> 5] = s;
    __syncthreads();
    float tot = 0.f;
    #pragma unroll
    for (int i = 0; i < 8; ++i) tot += sh[i];
    float mu = tot * (1.f / 512.f);
    __syncthreads();
    float d0 = v0 - mu, d1 = v1 - mu;
    float q = d0 * d0 + d1 * d1;
    #pragma unroll
    for (int o = 16; o; o >>= 1) q += __shfl_xor(q, o, 32);
    if ((tid & 31) == 0) sh[tid >> 5] = q;
    __syncthreads();
    float vtot = 0.f;
    #pragma unroll
    for (int i = 0; i < 8; ++i) vtot += sh[i];
    float rs = rsqrtf(vtot * (1.f / 512.f) + 1e-5f);
    out[(size_t)row * 512 + tid]       = f2bf(d0 * rs * gamma[tid]       + beta[tid]);
    out[(size_t)row * 512 + tid + 256] = f2bf(d1 * rs * gamma[tid + 256] + beta[tid + 256]);
}

// ---------------- bf16 WMMA GEMM ----------------
// C[M,N] = act( A[M,K](bf16) @ B[K,N](bf16) + bias + residual )
// Block: 256 thr = 8 waves; tile 128(M) x 64(N); wave w owns rows w*16..+15,
// computes 4 N-tiles (one A-fragment feeds 4 WMMAs per K-step).
// Double-buffered LDS; A staged with async-to-LDS, B register-pipelined + transposed.
__global__ __launch_bounds__(256) void gemm_bf16_wmma(
    const unsigned short* __restrict__ A, const unsigned short* __restrict__ B,
    const float* __restrict__ bias, const float* __restrict__ residual,
    float* __restrict__ Cf, unsigned short* __restrict__ Cbf,
    int M, int N, int K, int relu)
{
    __shared__ alignas(16) unsigned short sA[2][128 * 32];   // [row][k]
    __shared__ alignas(16) unsigned short sBT[2][64 * 32];   // transposed [n][k]
    const int tid  = threadIdx.x;
    const int wave = tid >> 5, lane = tid & 31;
    const int half = lane >> 4, l16 = lane & 15;
    const int row0 = blockIdx.x * 128, col0 = blockIdx.y * 64;

    auto stageA = [&](int p, int k0) {          // 128x32 tile: 512 x 16B, 2 per thread
        #pragma unroll
        for (int j = 0; j < 2; ++j) {
            int s = tid + j * 256;
            int r = s >> 2, seg = s & 3;
            copy16_g2l(A + (size_t)(row0 + r) * K + k0 + seg * 8,
                       &sA[p][r * 32 + seg * 8]);
        }
    };
    auto loadB = [&](unsigned int* bw, int k0) { // 32x64 tile as 1024 dwords, 4 per thread
        #pragma unroll
        for (int j = 0; j < 4; ++j) {
            int s = tid + j * 256;
            int kk = s >> 5, n2 = (s & 31) * 2;
            bw[j] = *(const unsigned int*)(B + (size_t)(k0 + kk) * N + col0 + n2);
        }
    };
    auto storeB = [&](const unsigned int* bw, int p) {
        #pragma unroll
        for (int j = 0; j < 4; ++j) {
            int s = tid + j * 256;
            int kk = s >> 5, n2 = (s & 31) * 2;
            sBT[p][n2 * 32 + kk]       = (unsigned short)(bw[j] & 0xffffu);
            sBT[p][(n2 + 1) * 32 + kk] = (unsigned short)(bw[j] >> 16);
        }
    };

    v8f acc[4];
    #pragma unroll
    for (int i = 0; i < 4; ++i) {
        v8f z = {0.f, 0.f, 0.f, 0.f, 0.f, 0.f, 0.f, 0.f};
        acc[i] = z;
    }

    unsigned int bw[4];
    loadB(bw, 0);
    stageA(0, 0);
    storeB(bw, 0);
    wait_async_all();
    __syncthreads();

    int p = 0;
    for (int k0 = 0; k0 < K; k0 += 32) {
        const bool has_next = (k0 + 32) < K;
        if (has_next) { loadB(bw, k0 + 32); stageA(p ^ 1, k0 + 32); }  // overlap w/ compute

        FragBF a;
        const unsigned short* ap = &sA[p][(wave * 16 + l16) * 32];
        #pragma unroll
        for (int g = 0; g < 8; ++g)
            a.u[g] = *(const unsigned int*)(ap + (g >> 2) * 16 + half * 8 + (g & 3) * 2);
        #pragma unroll
        for (int nt = 0; nt < 4; ++nt) {
            FragBF b;
            const unsigned short* bp = &sBT[p][(nt * 16 + l16) * 32];
            #pragma unroll
            for (int g = 0; g < 8; ++g)
                b.u[g] = *(const unsigned int*)(bp + (g >> 2) * 16 + half * 8 + (g & 3) * 2);
            acc[nt] = __builtin_amdgcn_wmma_f32_16x16x32_bf16(false, a.v, false, b.v,
                                                              (short)0, acc[nt], false, false);
        }
        if (has_next) { storeB(bw, p ^ 1); wait_async_all(); }
        __syncthreads();
        p ^= 1;
    }

    const int crow0 = row0 + wave * 16 + half * 8;
    #pragma unroll
    for (int nt = 0; nt < 4; ++nt) {
        const int ccol = col0 + nt * 16 + l16;
        const float bv = bias ? bias[ccol] : 0.f;
        #pragma unroll
        for (int v = 0; v < 8; ++v) {
            float val = acc[nt][v] + bv;
            size_t idx = (size_t)(crow0 + v) * N + ccol;
            if (residual) val += residual[idx];
            if (relu)     val = fmaxf(val, 0.f);
            if (Cf)  Cf[idx]  = val;
            if (Cbf) Cbf[idx] = f2bf(val);
        }
    }
}

// ---------------- fused dense attention per (batch, head) ----------------
// qkv: [1024][1536] bf16 (q|k|v each 512 = 8 heads x 64). out: [1024][512] bf16.
__global__ __launch_bounds__(256) void attention_kernel(const unsigned short* __restrict__ qkv,
                                                        unsigned short* __restrict__ out) {
    __shared__ alignas(16) unsigned short sVT[64 * 128];      // V^T [dk][token]
    __shared__ alignas(16) unsigned short sP[8 * 16 * 128];   // per-wave P rows
    const int blk = blockIdx.x;            // 64 = 8 batches x 8 heads
    const int b = blk >> 3, h = blk & 7;
    const int tok0 = b * 128;
    const int tid = threadIdx.x, wave = tid >> 5, lane = tid & 31;
    const int half = lane >> 4, l16 = lane & 15;

    // stage V transposed: 128 tok x 64 dk
    for (int s = tid; s < 4096; s += 256) {
        int tok = s >> 5, d = (s & 31) * 2;
        unsigned int w = *(const unsigned int*)(qkv + (size_t)(tok0 + tok) * 1536 + 1024 + h * 64 + d);
        sVT[d * 128 + tok]       = (unsigned short)(w & 0xffffu);
        sVT[(d + 1) * 128 + tok] = (unsigned short)(w >> 16);
    }
    __syncthreads();

    // Q fragments for this wave's 16 query rows (K-dim 64 = 2 steps of 32)
    FragBF qf[2];
    const unsigned short* qp = qkv + (size_t)(tok0 + wave * 16 + l16) * 1536 + h * 64;
    #pragma unroll
    for (int ks = 0; ks < 2; ++ks)
        #pragma unroll
        for (int g = 0; g < 8; ++g) {
            int kk = ks * 32 + (g >> 2) * 16 + half * 8 + (g & 3) * 2;
            qf[ks].u[g] = *(const unsigned int*)(qp + kk);
        }

    // S = Q K^T (8 key tiles). B-frag of K^T walks along K's contiguous row.
    v8f sc[8];
    #pragma unroll
    for (int nt = 0; nt < 8; ++nt) {
        v8f a = {0.f, 0.f, 0.f, 0.f, 0.f, 0.f, 0.f, 0.f};
        const unsigned short* kp = qkv + (size_t)(tok0 + nt * 16 + l16) * 1536 + 512 + h * 64;
        #pragma unroll
        for (int ks = 0; ks < 2; ++ks) {
            FragBF kf;
            #pragma unroll
            for (int g = 0; g < 8; ++g) kf.u[g] = *(const unsigned int*)(kp + ks * 32 + (g >> 2) * 16 + half * 8 + (g & 3) * 2);
            a = __builtin_amdgcn_wmma_f32_16x16x32_bf16(false, qf[ks].v, false, kf.v,
                                                        (short)0, a, false, false);
        }
        sc[nt] = a;
    }

    // row softmax: row r = v + 8*half lives in one 16-lane group, col = l16 + 16*nt
    unsigned short* P = &sP[wave * 16 * 128];
    const float scale = 0.125f;                 // 1/sqrt(64)
    #pragma unroll
    for (int v = 0; v < 8; ++v) {
        float m = -3.4e38f;
        #pragma unroll
        for (int nt = 0; nt < 8; ++nt) m = fmaxf(m, sc[nt][v] * scale);
        #pragma unroll
        for (int o = 8; o; o >>= 1) m = fmaxf(m, __shfl_xor(m, o, 32));
        float ev[8], z = 0.f;
        #pragma unroll
        for (int nt = 0; nt < 8; ++nt) { ev[nt] = __expf(sc[nt][v] * scale - m); z += ev[nt]; }
        #pragma unroll
        for (int o = 8; o; o >>= 1) z += __shfl_xor(z, o, 32);
        float inv = 1.f / z;
        int r = v + 8 * half;
        #pragma unroll
        for (int nt = 0; nt < 8; ++nt) P[r * 128 + nt * 16 + l16] = f2bf(ev[nt] * inv);
    }
    __syncthreads();

    // A = P V  (M=16, N=64 in 4 tiles, K=128 in 4 steps)
    #pragma unroll
    for (int nt2 = 0; nt2 < 4; ++nt2) {
        v8f a = {0.f, 0.f, 0.f, 0.f, 0.f, 0.f, 0.f, 0.f};
        #pragma unroll
        for (int ks = 0; ks < 4; ++ks) {
            FragBF pa, vb;
            const unsigned short* pp = P + l16 * 128 + ks * 32;
            #pragma unroll
            for (int g = 0; g < 8; ++g) {
                int kk = (g >> 2) * 16 + half * 8 + (g & 3) * 2;
                pa.u[g] = *(const unsigned int*)(pp + kk);
            }
            const unsigned short* vp = &sVT[(nt2 * 16 + l16) * 128 + ks * 32];
            #pragma unroll
            for (int g = 0; g < 8; ++g) vb.u[g] = *(const unsigned int*)(vp + (g >> 2) * 16 + half * 8 + (g & 3) * 2);
            a = __builtin_amdgcn_wmma_f32_16x16x32_bf16(false, pa.v, false, vb.v,
                                                        (short)0, a, false, false);
        }
        #pragma unroll
        for (int v = 0; v < 8; ++v) {
            int tok = tok0 + wave * 16 + v + 8 * half;
            out[(size_t)tok * 512 + h * 64 + nt2 * 16 + l16] = f2bf(a[v]);
        }
    }
}

// ---------------- row log-softmax in place ----------------
__global__ __launch_bounds__(256) void logsoftmax_rows(float* __restrict__ x, int N) {
    __shared__ float sh[8];
    int row = blockIdx.x, tid = threadIdx.x;
    float* p = x + (size_t)row * N;
    float m = -3.4e38f;
    for (int i = tid; i < N; i += 256) m = fmaxf(m, p[i]);
    #pragma unroll
    for (int o = 16; o; o >>= 1) m = fmaxf(m, __shfl_xor(m, o, 32));
    if ((tid & 31) == 0) sh[tid >> 5] = m;
    __syncthreads();
    float mm = sh[0];
    #pragma unroll
    for (int i = 1; i < 8; ++i) mm = fmaxf(mm, sh[i]);
    __syncthreads();
    float s = 0.f;
    for (int i = tid; i < N; i += 256) s += __expf(p[i] - mm);
    #pragma unroll
    for (int o = 16; o; o >>= 1) s += __shfl_xor(s, o, 32);
    if ((tid & 31) == 0) sh[tid >> 5] = s;
    __syncthreads();
    float z = 0.f;
    #pragma unroll
    for (int i = 0; i < 8; ++i) z += sh[i];
    float lz = logf(z);
    for (int i = tid; i < N; i += 256) p[i] = p[i] - mm - lz;
}

// ---------------- host side ----------------
extern "C" void kernel_launch(void* const* d_in, const int* in_sizes, int n_in,
                              void* d_out, int out_size, void* d_ws, size_t ws_size,
                              hipStream_t stream) {
    (void)in_sizes; (void)n_in; (void)out_size; (void)ws_size;
    const int*   tokens    = (const int*)d_in[0];
    const int*   positions = (const int*)d_in[1];
    const float* emb       = (const float*)d_in[4];
    const float* pos_table = (const float*)d_in[5];
    const float* ln1_g     = (const float*)d_in[6];
    const float* ln1_b     = (const float*)d_in[7];
    const float* W_qkv     = (const float*)d_in[8];
    const float* b_qkv     = (const float*)d_in[9];
    const float* W_o       = (const float*)d_in[10];
    const float* b_o       = (const float*)d_in[11];
    const float* ln2_g     = (const float*)d_in[12];
    const float* ln2_b     = (const float*)d_in[13];
    const float* W1        = (const float*)d_in[14];
    const float* b1        = (const float*)d_in[15];
    const float* W2        = (const float*)d_in[16];
    const float* b2        = (const float*)d_in[17];
    const float* gen_W     = (const float*)d_in[18];
    const float* gen_b     = (const float*)d_in[19];

    char* ws = (char*)d_ws;
    size_t off = 0;
    auto alloc = [&](size_t bytes) -> char* {
        char* p = ws + off;
        off = (off + bytes + 255) & ~(size_t)255;
        return p;
    };
    unsigned short* Wqkv_bf = (unsigned short*)alloc((size_t)4 * 512 * 1536 * 2);
    unsigned short* Wo_bf   = (unsigned short*)alloc((size_t)4 * 512 * 512 * 2);
    unsigned short* W1_bf   = (unsigned short*)alloc((size_t)4 * 512 * 2048 * 2);
    unsigned short* W2_bf   = (unsigned short*)alloc((size_t)4 * 2048 * 512 * 2);
    unsigned short* genW_bf = (unsigned short*)alloc((size_t)512 * 32000 * 2);
    float*          x       = (float*)alloc((size_t)1024 * 512 * 4);
    unsigned short* xn_bf   = (unsigned short*)alloc((size_t)1024 * 512 * 2);
    unsigned short* qkv_bf  = (unsigned short*)alloc((size_t)1024 * 1536 * 2);
    unsigned short* attn_bf = (unsigned short*)alloc((size_t)1024 * 512 * 2);
    unsigned short* h_bf    = (unsigned short*)alloc((size_t)1024 * 2048 * 2);
    unsigned short* x_bf    = (unsigned short*)alloc((size_t)1024 * 512 * 2);

    auto cgrid = [](int n) { return (n + 255) / 256; };
    conv_f32_bf16<<<cgrid(4 * 512 * 1536), 256, 0, stream>>>(W_qkv, Wqkv_bf, 4 * 512 * 1536);
    conv_f32_bf16<<<cgrid(4 * 512 * 512),  256, 0, stream>>>(W_o,   Wo_bf,   4 * 512 * 512);
    conv_f32_bf16<<<cgrid(4 * 512 * 2048), 256, 0, stream>>>(W1,    W1_bf,   4 * 512 * 2048);
    conv_f32_bf16<<<cgrid(4 * 2048 * 512), 256, 0, stream>>>(W2,    W2_bf,   4 * 2048 * 512);
    conv_f32_bf16<<<cgrid(512 * 32000),    256, 0, stream>>>(gen_W, genW_bf, 512 * 32000);

    embed_kernel<<<2048, 256, 0, stream>>>(tokens, positions, emb, pos_table, x);

    for (int l = 0; l < 4; ++l) {
        layernorm_bf16<<<1024, 256, 0, stream>>>(x, ln1_g + l * 512, ln1_b + l * 512, xn_bf);
        gemm_bf16_wmma<<<dim3(8, 24), 256, 0, stream>>>(
            xn_bf, Wqkv_bf + (size_t)l * 512 * 1536, b_qkv + l * 1536,
            nullptr, nullptr, qkv_bf, 1024, 1536, 512, 0);
        attention_kernel<<<64, 256, 0, stream>>>(qkv_bf, attn_bf);
        gemm_bf16_wmma<<<dim3(8, 8), 256, 0, stream>>>(
            attn_bf, Wo_bf + (size_t)l * 512 * 512, b_o + l * 512,
            x, x, nullptr, 1024, 512, 512, 0);
        layernorm_bf16<<<1024, 256, 0, stream>>>(x, ln2_g + l * 512, ln2_b + l * 512, xn_bf);
        gemm_bf16_wmma<<<dim3(8, 32), 256, 0, stream>>>(
            xn_bf, W1_bf + (size_t)l * 512 * 2048, b1 + l * 2048,
            nullptr, nullptr, h_bf, 1024, 2048, 512, 1);
        gemm_bf16_wmma<<<dim3(8, 8), 256, 0, stream>>>(
            h_bf, W2_bf + (size_t)l * 2048 * 512, b2 + l * 512,
            x, x, nullptr, 1024, 512, 2048, 0);
    }

    conv_f32_bf16<<<cgrid(1024 * 512), 256, 0, stream>>>(x, x_bf, 1024 * 512);
    gemm_bf16_wmma<<<dim3(8, 500), 256, 0, stream>>>(
        x_bf, genW_bf, gen_b, nullptr, (float*)d_out, nullptr, 1024, 32000, 512, 0);
    logsoftmax_rows<<<1024, 256, 0, stream>>>((float*)d_out, 32000);
}
